// FinalLayer_21062519619823
// MI455X (gfx1250) — matrix operand order repaired and checked by reading, not compiled, not run on hardware
//
#include <hip/hip_runtime.h>

#define B_    32
#define S_    1536
#define D_    2304
#define TWO_D 4608
#define NOUT  64      // P*P*C
#define L_IMG 1024
#define EPS_  1e-6f

#define AP 72         // padded LDS row (bf16 units): 144B, 16B-aligned, odd bank stride

typedef __attribute__((ext_vector_type(16))) __bf16 v16bf;
typedef __attribute__((ext_vector_type(8)))  float  v8f;
typedef __attribute__((ext_vector_type(4)))  unsigned int u32x4;

union Frag16 { v16bf v; u32x4 q[2]; };
union PkBf   { __bf16 h[2]; unsigned int u; };

// ---------------------------------------------------------------------------
// Kernel 1: mod = silu(adaln_input) @ w_adaln + b_adaln   [32, 4608] fp32
// One streaming pass over w_adaln (42.5 MB) -> memory-bound, ~2 us.
// ---------------------------------------------------------------------------
__global__ __launch_bounds__(256) void adaln_mod_kernel(
    const float* __restrict__ adaln, const float* __restrict__ w_adaln,
    const float* __restrict__ b_adaln, float* __restrict__ mod) {
  __shared__ float sl[64][32];  // [k][b] silu tile
  const int tid = threadIdx.x;
  const int n = blockIdx.x * 256 + tid;
  float acc[32];
#pragma unroll
  for (int b = 0; b < 32; ++b) acc[b] = 0.f;

  for (int k0 = 0; k0 < D_; k0 += 64) {
    __syncthreads();
#pragma unroll
    for (int i = 0; i < 8; ++i) {
      int e = tid + i * 256;       // 0..2047
      int kk = e & 63;
      int bb = e >> 6;
      float v = adaln[bb * D_ + k0 + kk];
      sl[kk][bb] = v / (1.f + __expf(-v));
    }
    __syncthreads();
    for (int kk = 0; kk < 64; ++kk) {
      float w = w_adaln[(size_t)(k0 + kk) * TWO_D + n];
#pragma unroll
      for (int b4 = 0; b4 < 8; ++b4) {
        float4 s4 = *(const float4*)&sl[kk][b4 * 4];
        acc[b4 * 4 + 0] += s4.x * w;
        acc[b4 * 4 + 1] += s4.y * w;
        acc[b4 * 4 + 2] += s4.z * w;
        acc[b4 * 4 + 3] += s4.w * w;
      }
    }
  }
  float bias = b_adaln[n];
#pragma unroll
  for (int b = 0; b < 32; ++b) mod[(size_t)b * TWO_D + n] = acc[b] + bias;
}

// ---------------------------------------------------------------------------
// Kernel 1b: w_lin fp32 -> bf16 (288 KB, stays L2-hot for kernel 2)
// ---------------------------------------------------------------------------
__global__ void cvt_wlin_kernel(const float* __restrict__ w,
                                __bf16* __restrict__ o, int n) {
  int i = blockIdx.x * 256 + threadIdx.x;
  if (i < n) o[i] = (__bf16)w[i];
}

__global__ void zero_out_kernel(float* o) {
  if (threadIdx.x == 0) o[0] = 0.f;
}

// ---------------------------------------------------------------------------
// Kernel 2: fused LN + modulate + GEMM(WMMA bf16) + unpatchify + MSE
// grid (8 tiles, 32 samples), 256 threads = 8 waves; wave -> 16x64 out tile.
// K-chunk = 64: 8 v_wmma per wave between barriers; staging packed to
// b64 global loads + b32 DS stores.
// ---------------------------------------------------------------------------
__global__ __launch_bounds__(256, 2) void final_layer_kernel(
    const float* __restrict__ x, const int* __restrict__ cap_size,
    const float* __restrict__ target, const float* __restrict__ mod,
    const __bf16* __restrict__ wlin, const float* __restrict__ b_lin,
    float* __restrict__ out) {
  __shared__ __align__(16) __bf16 Alds[128][AP];  // tokens x k (18 KB)
  __shared__ __align__(16) __bf16 Blds[64][AP];   // n x k, transposed (9 KB)
  __shared__ float muS[128], isS[128];
  __shared__ float redS[8];

  const int tid  = threadIdx.x;
  const int lane = tid & 31;
  const int wave = tid >> 5;
  const int b    = blockIdx.y;
  const int l0   = blockIdx.x * 128;           // first image token of block
  const int s0   = cap_size[b] + l0;

  const float* xb    = x + ((size_t)b * S_ + s0) * D_;
  const float* modsh = mod + (size_t)b * TWO_D;   // shift row
  const float* modsc = modsh + D_;                // scale row

  // ---- Phase 1: LayerNorm stats; wave w -> tokens 16w..16w+15 (float4 loads)
  for (int tt = 0; tt < 16; ++tt) {
    int t = wave * 16 + tt;
    const float4* row = (const float4*)(xb + (size_t)t * D_);
    float s = 0.f, s2 = 0.f;
#pragma unroll 3
    for (int i = 0; i < D_ / 128; ++i) {       // 18 iters
      float4 v = row[lane + i * 32];
      s  += v.x + v.y + v.z + v.w;
      s2 += v.x * v.x + v.y * v.y + v.z * v.z + v.w * v.w;
    }
#pragma unroll
    for (int off = 16; off > 0; off >>= 1) {
      s  += __shfl_xor(s, off);
      s2 += __shfl_xor(s2, off);
    }
    if (lane == 0) {
      float mu  = s * (1.f / D_);
      float var = s2 * (1.f / D_) - mu * mu;
      muS[t] = mu;
      isS[t] = rsqrtf(var + EPS_);
    }
  }
  __syncthreads();

  // ---- Phase 2: K-loop, 36 chunks of 64
  v8f c[4] = {v8f{}, v8f{}, v8f{}, v8f{}};
  const int kj = (tid & 31) * 2;   // even k within chunk (A staging pair)
  const int tr = tid >> 5;         // token base (A staging), 0..7
  for (int k0 = 0; k0 < D_; k0 += 64) {
    float2 sh = *(const float2*)&modsh[k0 + kj];
    float2 sc2 = *(const float2*)&modsc[k0 + kj];
    float scx = 1.f + sc2.x, scy = 1.f + sc2.y;
    // stage A: normalized + modulated tokens, packed bf16 pairs
#pragma unroll
    for (int i = 0; i < 16; ++i) {
      int t = tr + i * 8;
      float2 v = *(const float2*)&xb[(size_t)t * D_ + k0 + kj];
      float ms = muS[t], is = isS[t];
      PkBf p;
      p.h[0] = (__bf16)((v.x - ms) * is * scx + sh.x);
      p.h[1] = (__bf16)((v.y - ms) * is * scy + sh.y);
      *(unsigned int*)&Alds[t][kj] = p.u;
    }
    // stage B transposed, packed pairs: Blds[n][k]
#pragma unroll
    for (int i = 0; i < 8; ++i) {
      int e  = tid + i * 256;    // 0..2047 pair-slots
      int n  = e & 63;
      int kr = (e >> 6) * 2;     // 0,2,..,62
      PkBf p;
      p.h[0] = wlin[(size_t)(k0 + kr) * NOUT + n];
      p.h[1] = wlin[(size_t)(k0 + kr + 1) * NOUT + n];
      *(unsigned int*)&Blds[n][kr] = p.u;
    }
    __syncthreads();

    // fragments per ISA 16-bit layouts (two K=32 halves)
    const int row = lane & 15;
    const int kb  = (lane >> 4) * 8;    // A: K 0-7/16-23 vs 8-15/24-31
    const int klo = (lane >> 4) * 16;
    const __bf16* ar = &Alds[wave * 16 + row][0];
    Frag16 a0, a1;
    a0.q[0] = *(const u32x4*)&ar[kb];
    a0.q[1] = *(const u32x4*)&ar[16 + kb];
    a1.q[0] = *(const u32x4*)&ar[32 + kb];
    a1.q[1] = *(const u32x4*)&ar[48 + kb];
#pragma unroll
    for (int nt = 0; nt < 4; ++nt) {
      const __bf16* br = &Blds[nt * 16 + row][0];
      Frag16 b0, b1;
      b0.q[0] = *(const u32x4*)&br[kb];
      b0.q[1] = *(const u32x4*)&br[16 + kb];
      b1.q[0] = *(const u32x4*)&br[32 + kb];
      b1.q[1] = *(const u32x4*)&br[48 + kb];
      c[nt] = __builtin_amdgcn_wmma_f32_16x16x32_bf16(
          false, a0.v, false, b0.v, (short)0, c[nt], false, false);
      c[nt] = __builtin_amdgcn_wmma_f32_16x16x32_bf16(
          false, a1.v, false, b1.v, (short)0, c[nt], false, false);
    }
    __syncthreads();
  }

  // ---- Epilogue: bias + unpatchify-indexed squared error, fused reduction
  float mse = 0.f;
  const int colb = lane & 15;
  const int mhi  = lane >> 4;
  const float* tb = target + (size_t)b * (16 * 64 * 64);
#pragma unroll
  for (int nt = 0; nt < 4; ++nt) {
    int n  = nt * 16 + colb;
    float bl = b_lin[n];
    int p1 = n >> 5, p2 = (n >> 4) & 1, ch = n & 15;
#pragma unroll
    for (int r = 0; r < 8; ++r) {
      int m  = r + 8 * mhi;
      int l  = l0 + wave * 16 + m;
      int hp = l >> 5, wp = l & 31;
      int h  = hp * 2 + p1, w = wp * 2 + p2;
      float val = c[nt][r] + bl;
      float d = val - tb[((size_t)ch * 64 + h) * 64 + w];
      mse += d * d;
    }
  }
#pragma unroll
  for (int off = 16; off > 0; off >>= 1) mse += __shfl_xor(mse, off);
  if (lane == 0) redS[wave] = mse;
  __syncthreads();
  if (tid == 0) {
    float t = 0.f;
#pragma unroll
    for (int i = 0; i < 8; ++i) t += redS[i];
    atomicAdd(out, t * (1.f / (32.f * 16.f * 64.f * 64.f)));
  }
}

// ---------------------------------------------------------------------------
extern "C" void kernel_launch(void* const* d_in, const int* in_sizes, int n_in,
                              void* d_out, int out_size, void* d_ws,
                              size_t ws_size, hipStream_t stream) {
  const float* x        = (const float*)d_in[0];
  const float* adaln    = (const float*)d_in[1];
  const int*   cap      = (const int*)d_in[2];
  const float* target   = (const float*)d_in[3];
  const float* w_adaln  = (const float*)d_in[4];
  const float* b_adaln  = (const float*)d_in[5];
  const float* w_lin    = (const float*)d_in[6];
  const float* b_lin    = (const float*)d_in[7];
  float* out = (float*)d_out;

  float*  mod   = (float*)d_ws;                                    // 32*4608 f32
  __bf16* wlinb = (__bf16*)((char*)d_ws + (size_t)B_ * TWO_D * 4); // 2304*64 bf16

  adaln_mod_kernel<<<dim3(TWO_D / 256), 256, 0, stream>>>(adaln, w_adaln,
                                                          b_adaln, mod);
  cvt_wlin_kernel<<<dim3((D_ * NOUT + 255) / 256), 256, 0, stream>>>(
      w_lin, wlinb, D_ * NOUT);
  zero_out_kernel<<<1, 32, 0, stream>>>(out);
  final_layer_kernel<<<dim3(8, 32), 256, 0, stream>>>(x, cap, target, mod,
                                                      wlinb, b_lin, out);
}
